// DGCNN_36988258353641
// MI455X (gfx1250) — compile-verified, matrix operand order checked by reading, not run on hardware
//
#include <hip/hip_runtime.h>

// ---------------------------------------------------------------------------
// DGCNN forward for MI455X (gfx1250, wave32). Dense GEMMs use
// v_wmma_f32_16x16x32_bf16 (bf16 in, f32 accumulate) with one wave owning a
// full row of N-tiles. Per k-step: all fragment loads are issued as one clause
// into distinct register arrays, then NT independent WMMAs run back-to-back
// (no D->A/B hazards, partial loadcnt waits). GCN self-loop term fused into
// the GEMM epilogue. Edge aggregation uses L2-resident float atomics.
// ---------------------------------------------------------------------------

#define NN    65536
#define EE    1048576
#define GG    512
#define PP    128
#define KTOP  64
#define CCAT  256

typedef __attribute__((ext_vector_type(16))) __bf16 bfx16;
typedef __attribute__((ext_vector_type(8)))  __bf16 bfx8;
typedef __attribute__((ext_vector_type(8)))  float  fx8;

static inline int cdiv_l(long a, int b) { return (int)((a + b - 1) / b); }

// ---------------- elementwise / prep kernels ----------------

__global__ void k_f32_to_bf16(const float* __restrict__ s, __bf16* __restrict__ d, long n) {
    long i = (long)blockIdx.x * blockDim.x + threadIdx.x;
    if (i < n) d[i] = (__bf16)s[i];
}

// W: [K x N] row-major f32  ->  WT: [N x K] row-major bf16 (i.e. B column-major)
__global__ void k_transpose_bf16(const float* __restrict__ W, __bf16* __restrict__ WT,
                                 int K, int N) {
    long i = (long)blockIdx.x * blockDim.x + threadIdx.x;
    long tot = (long)K * N;
    if (i >= tot) return;
    int k = (int)(i / N);
    int n = (int)(i - (long)k * N);
    WT[(size_t)n * K + k] = (__bf16)W[i];
}

__global__ void k_fill(float* __restrict__ p, float v, long n) {
    long i = (long)blockIdx.x * blockDim.x + threadIdx.x;
    if (i < n) p[i] = v;
}

__global__ void k_deg(const int* __restrict__ dst, float* __restrict__ deg, int E) {
    long i = (long)blockIdx.x * blockDim.x + threadIdx.x;
    if (i < E) atomicAdd(&deg[dst[i]], 1.0f);
}

__global__ void k_rsqrt_inplace(float* __restrict__ d, long n) {
    long i = (long)blockIdx.x * blockDim.x + threadIdx.x;
    if (i < n) d[i] = rsqrtf(d[i]);
}

// ---------------- WMMA bf16 GEMM, one wave per M-tile row ----------------
// C[M x Nout] = A[M x K] * B[K x Nout], Nout = NT*16.
// A  : row-major bf16, leading dim lda
// BT : B stored N-major: BT[n*K + k] = B[k][n], bf16
// C  : f32, leading dim ldc (= Nout here)
// Optional fused GCN self-loop epilogue: agg = C*dinv[row]^2 + bias[col].
// Requires: M % 16 == 0, K % 32 == 0, lda/K multiples of 8.
template<int NT>
__global__ void k_gemm_wmma_rowtile(const __bf16* __restrict__ A, int lda,
                                    const __bf16* __restrict__ BT,
                                    float* __restrict__ C, int ldc,
                                    int M, int K,
                                    float* __restrict__ agg,
                                    const float* __restrict__ dinv,
                                    const float* __restrict__ bias) {
    const int wave = threadIdx.x >> 5;
    const int lane = threadIdx.x & 31;
    const int tm = blockIdx.x * (blockDim.x >> 5) + wave;
    if (tm >= (M >> 4)) return;               // wave-uniform exit, EXEC stays all-1s

    const int l15 = lane & 15;
    const int hi8 = (lane >> 4) << 3;         // 0 for lanes 0-15, 8 for lanes 16-31

    const __bf16* arow = A + (size_t)(tm * 16 + l15) * lda + hi8;
    const __bf16* bptr[NT];
#pragma unroll
    for (int nt = 0; nt < NT; ++nt)
        bptr[nt] = BT + (size_t)(nt * 16 + l15) * K + hi8;

    fx8 acc[NT] = {};
    for (int kc = 0; kc < K; kc += 32) {
        // Issue the whole fragment-load set as one clause, then run NT
        // independent WMMAs (no D->A/B deps; scheduler can partial-wait).
        __builtin_prefetch(arow + kc + 256, 0, 3);   // global_prefetch_b8
        bfx8 a0 = *(const bfx8*)(arow + kc);
        bfx8 a1 = *(const bfx8*)(arow + kc + 16);
        bfx8 b0[NT], b1[NT];
#pragma unroll
        for (int nt = 0; nt < NT; ++nt) {
            b0[nt] = *(const bfx8*)(bptr[nt] + kc);
            b1[nt] = *(const bfx8*)(bptr[nt] + kc + 16);
        }
        bfx16 av;
#pragma unroll
        for (int i = 0; i < 8; ++i) { av[i] = a0[i]; av[i + 8] = a1[i]; }
#pragma unroll
        for (int nt = 0; nt < NT; ++nt) {
            bfx16 bv;
#pragma unroll
            for (int i = 0; i < 8; ++i) { bv[i] = b0[nt][i]; bv[i + 8] = b1[nt][i]; }
            acc[nt] = __builtin_amdgcn_wmma_f32_16x16x32_bf16(
                false, av, false, bv, (short)0, acc[nt], false, false);
        }
    }

    const int mb = tm * 16 + hi8;
    float bs[NT];
    if (agg) {
#pragma unroll
        for (int nt = 0; nt < NT; ++nt) bs[nt] = bias[nt * 16 + l15];
    }
#pragma unroll
    for (int r = 0; r < 8; ++r) {
        const int row = mb + r;
        float di2 = 0.0f;
        if (agg) { float di = dinv[row]; di2 = di * di; }
        float* crow = C + (size_t)row * ldc + l15;
        float* arow2 = agg ? agg + (size_t)row * ldc + l15 : nullptr;
#pragma unroll
        for (int nt = 0; nt < NT; ++nt) {
            float v = acc[nt][r];
            crow[nt * 16] = v;
            if (agg) arow2[nt * 16] = v * di2 + bs[nt];
        }
    }
}

// ---------------- GCN aggregation ----------------

// agg[dst,c] += H[src,c] * dinv[src]*dinv[dst], vectorized 4 channels/thread
__global__ void k_gcn_scatter(const float* __restrict__ H, const float* __restrict__ dinv,
                              const int* __restrict__ src, const int* __restrict__ dst,
                              float* __restrict__ agg, int E, int C) {
    const int Cq = C >> 2;
    long i = (long)blockIdx.x * blockDim.x + threadIdx.x;
    long tot = (long)E * Cq;
    if (i >= tot) return;
    int e = (int)(i / Cq);
    int q = (int)(i - (long)e * Cq);
    int s = src[e], d = dst[e];
    float coef = dinv[s] * dinv[d];
    const float4 h4 = *(const float4*)(H + (size_t)s * C + (q << 2));
    float* ap = agg + (size_t)d * C + (q << 2);
    atomicAdd(ap + 0, h4.x * coef);
    atomicAdd(ap + 1, h4.y * coef);
    atomicAdd(ap + 2, h4.z * coef);
    atomicAdd(ap + 3, h4.w * coef);
}

// x_out = tanh(agg); write f32 into hcat slice + compact bf16 for next GEMM
__global__ void k_gcn_act(const float* __restrict__ agg, float* __restrict__ hcat,
                          int coloff, __bf16* __restrict__ xb, int Nn, int C) {
    long i = (long)blockIdx.x * blockDim.x + threadIdx.x;
    long tot = (long)Nn * C;
    if (i >= tot) return;
    int n = (int)(i / C);
    int c = (int)(i - (long)n * C);
    float v = tanhf(agg[i]);
    hcat[(size_t)n * CCAT + coloff + c] = v;
    xb[i] = (__bf16)v;
}

// ---------------- sort pool: per-graph bitonic top-K gather ----------------
// one block (128 threads) per graph; descending sort by hcat[:,255], stable ties.
__global__ void k_sortpool(const float* __restrict__ hcat, __bf16* __restrict__ pooledb) {
    __shared__ float skey[PP];
    __shared__ int   sidx[PP];
    const int g = blockIdx.x;
    const int t = threadIdx.x;

    skey[t] = hcat[((size_t)(g * PP + t)) * CCAT + (CCAT - 1)];
    sidx[t] = t;
    __syncthreads();

    for (int ks = 2; ks <= PP; ks <<= 1) {
        for (int j = ks >> 1; j > 0; j >>= 1) {
            int i = t, ixj = t ^ j;
            if (ixj > i) {
                bool desc = ((i & ks) == 0);
                float ki = skey[i], kj = skey[ixj];
                int ii = sidx[i], jj = sidx[ixj];
                bool swp = desc ? (ki < kj || (ki == kj && ii > jj))
                                : (ki > kj || (ki == kj && ii < jj));
                if (swp) { skey[i] = kj; skey[ixj] = ki; sidx[i] = jj; sidx[ixj] = ii; }
            }
            __syncthreads();
        }
    }

    for (int u = t; u < KTOP * CCAT; u += PP) {
        int r = u >> 8;           // / CCAT
        int c = u & (CCAT - 1);
        float v = hcat[((size_t)(g * PP + sidx[r])) * CCAT + c];
        pooledb[((size_t)(g * KTOP + r)) * CCAT + c] = (__bf16)v;
    }
}

// ---------------- post-pool head ----------------

// z: [G*K, 16] (row = g*64+t, col = c).  zp[g][c][t2] = relu(max(z(2t2), z(2t2+1)) + b5[c])
__global__ void k_zpool(const float* __restrict__ z, const float* __restrict__ b5,
                        float* __restrict__ zp) {
    long i = (long)blockIdx.x * blockDim.x + threadIdx.x;
    long tot = (long)GG * 16 * 32;
    if (i >= tot) return;
    int g  = (int)(i / (16 * 32));
    int r  = (int)(i - (long)g * 16 * 32);
    int c  = r >> 5;
    int t2 = r & 31;
    float v0 = z[((size_t)(g * KTOP + 2 * t2)) * 16 + c];
    float v1 = z[((size_t)(g * KTOP + 2 * t2 + 1)) * 16 + c];
    zp[i] = fmaxf(fmaxf(v0, v1) + b5[c], 0.0f);
}

// Conv1d(16,32,5) valid + relu, flatten (o*28+t); write f32 + bf16
__global__ void k_conv1d(const float* __restrict__ zp, const float* __restrict__ W6,
                         const float* __restrict__ b6, float* __restrict__ flat,
                         __bf16* __restrict__ flatb) {
    __shared__ float s_in[16 * 32];
    __shared__ float s_w[32 * 16 * 5];
    const int g = blockIdx.x;
    for (int i = threadIdx.x; i < 16 * 32; i += blockDim.x) s_in[i] = zp[g * 512 + i];
    for (int i = threadIdx.x; i < 32 * 16 * 5; i += blockDim.x) s_w[i] = W6[i];
    __syncthreads();
    for (int u = threadIdx.x; u < 32 * 28; u += blockDim.x) {
        int o = u / 28, t = u - o * 28;
        float acc = b6[o];
#pragma unroll
        for (int i = 0; i < 16; ++i)
#pragma unroll
            for (int tau = 0; tau < 5; ++tau)
                acc += s_in[i * 32 + t + tau] * s_w[(o * 16 + i) * 5 + tau];
        acc = fmaxf(acc, 0.0f);
        flat[(size_t)g * 896 + u]  = acc;
        flatb[(size_t)g * 896 + u] = (__bf16)acc;
    }
}

// in-place relu(x + bias[c])
__global__ void k_relu_bias(float* __restrict__ buf, const float* __restrict__ bias,
                            int M, int C) {
    long i = (long)blockIdx.x * blockDim.x + threadIdx.x;
    long tot = (long)M * C;
    if (i >= tot) return;
    int c = (int)(i % C);
    buf[i] = fmaxf(buf[i] + bias[c], 0.0f);
}

// FC2 (128 -> 10) + log_softmax; one wave per graph
__global__ void k_fc2_logsoftmax(const float* __restrict__ fc1, const float* __restrict__ Wf2,
                                 const float* __restrict__ bf2, float* __restrict__ out) {
    __shared__ float sv[10];
    __shared__ float sred[2];
    const int g = blockIdx.x;
    const int t = threadIdx.x;
    if (t < 10) {
        float acc = bf2[t];
        const float* row = fc1 + (size_t)g * 128;
        for (int k = 0; k < 128; ++k) acc += row[k] * Wf2[k * 10 + t];
        sv[t] = acc;
    }
    __syncthreads();
    if (t == 0) {
        float m = sv[0];
        for (int j = 1; j < 10; ++j) m = fmaxf(m, sv[j]);
        float s = 0.0f;
        for (int j = 0; j < 10; ++j) s += expf(sv[j] - m);
        sred[0] = m;
        sred[1] = logf(s);
    }
    __syncthreads();
    if (t < 10) out[(size_t)g * 10 + t] = sv[t] - sred[0] - sred[1];
}

// ---------------------------------------------------------------------------

extern "C" void kernel_launch(void* const* d_in, const int* in_sizes, int n_in,
                              void* d_out, int out_size, void* d_ws, size_t ws_size,
                              hipStream_t stream) {
    const float* x   = (const float*)d_in[0];
    const float* W1  = (const float*)d_in[1];  const float* b1  = (const float*)d_in[2];
    const float* W2  = (const float*)d_in[3];  const float* b2  = (const float*)d_in[4];
    const float* W3  = (const float*)d_in[5];  const float* b3  = (const float*)d_in[6];
    const float* W4  = (const float*)d_in[7];  const float* b4  = (const float*)d_in[8];
    const float* W5  = (const float*)d_in[9];  const float* b5  = (const float*)d_in[10];
    const float* W6  = (const float*)d_in[11]; const float* b6  = (const float*)d_in[12];
    const float* Wf1 = (const float*)d_in[13]; const float* bf1 = (const float*)d_in[14];
    const float* Wf2 = (const float*)d_in[15]; const float* bf2 = (const float*)d_in[16];
    const int*   ei  = (const int*)d_in[17];
    const int* srcp = ei;           // edge_index[0]
    const int* dstp = ei + EE;      // edge_index[1]
    float* outp = (float*)d_out;

    // ---- workspace carve (256B aligned regions) ----
    char* wp = (char*)d_ws;
    auto carve = [&](size_t bytes) -> void* {
        void* r = (void*)wp;
        wp += (bytes + 255) & ~(size_t)255;
        return r;
    };
    float*  dinv    = (float*) carve((size_t)NN * 4);
    __bf16* xb      = (__bf16*)carve((size_t)NN * 128 * 2);
    float*  H       = (float*) carve((size_t)NN * 128 * 4);
    float*  agg     = (float*) carve((size_t)NN * 128 * 4);
    float*  hcat    = (float*) carve((size_t)NN * CCAT * 4);
    __bf16* pooledb = (__bf16*)carve((size_t)GG * KTOP * CCAT * 2);
    float*  z       = (float*) carve((size_t)GG * KTOP * 16 * 4);
    float*  zp      = (float*) carve((size_t)GG * 16 * 32 * 4);
    float*  flat    = (float*) carve((size_t)GG * 896 * 4);
    __bf16* flatb   = (__bf16*)carve((size_t)GG * 896 * 2);
    float*  fc1     = (float*) carve((size_t)GG * 128 * 4);
    __bf16* W1T     = (__bf16*)carve(128 * 128 * 2);
    __bf16* W2T     = (__bf16*)carve(64  * 128 * 2);
    __bf16* W3T     = (__bf16*)carve(32  * 64  * 2);
    __bf16* W4T     = (__bf16*)carve(32  * 32  * 2);
    __bf16* W5b     = (__bf16*)carve(16  * 256 * 2);
    __bf16* Wf1T    = (__bf16*)carve(128 * 896 * 2);
    (void)ws_size; (void)in_sizes; (void)n_in; (void)out_size;

    const int B = 256;

    // ---- prep: bf16 conversions / weight transposes ----
    k_f32_to_bf16<<<cdiv_l((long)NN * 128, B), B, 0, stream>>>(x, xb, (long)NN * 128);
    k_transpose_bf16<<<cdiv_l(128 * 128, B), B, 0, stream>>>(W1,  W1T, 128, 128);
    k_transpose_bf16<<<cdiv_l(128 * 64,  B), B, 0, stream>>>(W2,  W2T, 128, 64);
    k_transpose_bf16<<<cdiv_l(64 * 32,   B), B, 0, stream>>>(W3,  W3T, 64,  32);
    k_transpose_bf16<<<cdiv_l(32 * 32,   B), B, 0, stream>>>(W4,  W4T, 32,  32);
    k_f32_to_bf16<<<cdiv_l(16 * 256, B), B, 0, stream>>>(W5, W5b, 16 * 256); // already [16,256]=N-major
    k_transpose_bf16<<<cdiv_l(896 * 128, B), B, 0, stream>>>(Wf1, Wf1T, 896, 128);

    // ---- degree / dinv ----
    k_fill<<<cdiv_l(NN, B), B, 0, stream>>>(dinv, 1.0f, NN);
    k_deg<<<cdiv_l(EE, B), B, 0, stream>>>(dstp, dinv, EE);
    k_rsqrt_inplace<<<cdiv_l(NN, B), B, 0, stream>>>(dinv, NN);

    // ---- GCN layers: GEMM (WMMA, fused self-loop) -> scatter -> tanh ----
    const int mtiles = NN / 16;
    const int gblk   = cdiv_l(mtiles, 8);
    // L1: 128 -> 128
    k_gemm_wmma_rowtile<8><<<gblk, B, 0, stream>>>(xb, 128, W1T, H, 128, NN, 128,
                                                   agg, dinv, b1);
    k_gcn_scatter<<<cdiv_l((long)EE * 32, B), B, 0, stream>>>(H, dinv, srcp, dstp, agg, EE, 128);
    k_gcn_act<<<cdiv_l((long)NN * 128, B), B, 0, stream>>>(agg, hcat, 0, xb, NN, 128);
    // L2: 128 -> 64
    k_gemm_wmma_rowtile<4><<<gblk, B, 0, stream>>>(xb, 128, W2T, H, 64, NN, 128,
                                                   agg, dinv, b2);
    k_gcn_scatter<<<cdiv_l((long)EE * 16, B), B, 0, stream>>>(H, dinv, srcp, dstp, agg, EE, 64);
    k_gcn_act<<<cdiv_l((long)NN * 64, B), B, 0, stream>>>(agg, hcat, 128, xb, NN, 64);
    // L3: 64 -> 32
    k_gemm_wmma_rowtile<2><<<gblk, B, 0, stream>>>(xb, 64, W3T, H, 32, NN, 64,
                                                   agg, dinv, b3);
    k_gcn_scatter<<<cdiv_l((long)EE * 8, B), B, 0, stream>>>(H, dinv, srcp, dstp, agg, EE, 32);
    k_gcn_act<<<cdiv_l((long)NN * 32, B), B, 0, stream>>>(agg, hcat, 192, xb, NN, 32);
    // L4: 32 -> 32
    k_gemm_wmma_rowtile<2><<<gblk, B, 0, stream>>>(xb, 32, W4T, H, 32, NN, 32,
                                                   agg, dinv, b4);
    k_gcn_scatter<<<cdiv_l((long)EE * 8, B), B, 0, stream>>>(H, dinv, srcp, dstp, agg, EE, 32);
    k_gcn_act<<<cdiv_l((long)NN * 32, B), B, 0, stream>>>(agg, hcat, 224, xb, NN, 32);

    // ---- sort pool (top-64 of 128 per graph) ----
    k_sortpool<<<GG, PP, 0, stream>>>(hcat, pooledb);

    // ---- Conv1d(1,16,256,256) == [G*K,256] x [256,16] GEMM ----
    k_gemm_wmma_rowtile<1><<<cdiv_l((GG * KTOP) / 16, 8), B, 0, stream>>>(
        pooledb, CCAT, W5b, z, 16, GG * KTOP, CCAT, nullptr, nullptr, nullptr);
    k_zpool<<<cdiv_l((long)GG * 16 * 32, B), B, 0, stream>>>(z, b5, zp);

    // ---- Conv1d(16,32,5) + relu + flatten ----
    k_conv1d<<<GG, B, 0, stream>>>(zp, W6, b6, flat, flatb);

    // ---- FC1: [512,896] x [896,128] (WMMA) + relu ----
    k_gemm_wmma_rowtile<8><<<cdiv_l(GG / 16, 8), B, 0, stream>>>(
        flatb, 896, Wf1T, fc1, 128, GG, 896, nullptr, nullptr, nullptr);
    k_relu_bias<<<cdiv_l((long)GG * 128, B), B, 0, stream>>>(fc1, bf1, GG, 128);

    // ---- FC2 + log_softmax ----
    k_fc2_logsoftmax<<<GG, 32, 0, stream>>>(fc1, Wf2, bf2, outp);
}
